// LSTM_43430709297555
// MI455X (gfx1250) — matrix operand order repaired and checked by reading, not compiled, not run on hardware
//
#include <hip/hip_runtime.h>
#include <hip/hip_bf16.h>
#include <math.h>

// ---------------------------------------------------------------- types
typedef __bf16 bf16_t;
typedef __attribute__((ext_vector_type(16))) __bf16 v16bf;
typedef __attribute__((ext_vector_type(8)))  __bf16 v8bf;
typedef __attribute__((ext_vector_type(8)))  float  v8f;

// ---------------------------------------------------------------- problem dims
#define BB   64
#define TT   512
#define TSH  9             // log2(TT)
#define II   256
#define HH   512
#define OO   256
#define G4H  2048          // 4*H
#define BT   32768         // B*T
#define HPAD 520           // H + 8 bf16 pad -> 1040B row, bank-conflict free

#define OUT_H_OFF  ((size_t)BT * OO)            // 8,388,608 floats
#define OUT_C_OFF  (OUT_H_OFF + (size_t)BB*HH)  // + 32,768

// ---------------------------------------------------------------- helpers
__device__ __forceinline__ v16bf cat16(v8bf a, v8bf b) {
  return __builtin_shufflevector(a, b, 0,1,2,3,4,5,6,7,8,9,10,11,12,13,14,15);
}

// A tile (16x32, MxK, bf16): lane holds row m=lane&15, 16 contiguous K values.
__device__ __forceinline__ v16bf load_A16(const bf16_t* base_m0, int ld, int kb, int lane) {
  int m  = lane & 15;
  int kh = (lane >> 4) << 4;          // 0 or 16
  const v8bf* p0 = (const v8bf*)(base_m0 + (size_t)m * ld + kb + kh);
  const v8bf* p1 = (const v8bf*)(base_m0 + (size_t)m * ld + kb + kh + 8);
  return cat16(*p0, *p1);
}

// B tile (32x16, KxN, bf16) from K-major (transposed) matrix Bt[N][K]:
// lane holds column n=lane&15, 16 contiguous K values.
__device__ __forceinline__ v16bf load_B16(const bf16_t* base_n0, int ld, int kb, int lane) {
  int n  = lane & 15;
  int ks = (lane >> 4) << 4;          // 0 or 16
  const v8bf* p = (const v8bf*)(base_n0 + (size_t)n * ld + kb + ks);
  return cat16(p[0], p[1]);
}

__device__ __forceinline__ v8f wmma_bf16(v16bf a, v16bf b, v8f c) {
  return __builtin_amdgcn_wmma_f32_16x16x32_bf16(false, a, false, b, (short)0, c, false, false);
}

__device__ __forceinline__ float sigmoidf_(float x) { return 1.0f / (1.0f + __expf(-x)); }

// ---------------------------------------------------------------- pack kernels
__global__ void pack_cast_bf16(const float* __restrict__ src, bf16_t* __restrict__ dst, int n) {
  for (int i = blockIdx.x * blockDim.x + threadIdx.x; i < n; i += gridDim.x * blockDim.x)
    dst[i] = (bf16_t)src[i];
}

// dst[C][R] = src[R][C]  (bf16 K-major operand for WMMA B)
__global__ void pack_transpose_bf16(const float* __restrict__ src, bf16_t* __restrict__ dst,
                                    int R, int C) {
  int i = blockIdx.x * blockDim.x + threadIdx.x;
  if (i < R * C) {
    int r = i / C, c = i % C;
    dst[(size_t)c * R + r] = (bf16_t)src[i];
  }
}

// ---------------------------------------------------------------- WMMA GEMM, 16x64 per wave
// C = A[M][K](bf16) * Bt[N][K](bf16)^T + bias[N]   (fp32 out)
// 2x-unrolled double-buffered pipeline: stage loads land in alternate named
// registers, so no register-rotation moves; K must be a multiple of 64.
// perm==0 : C row = m                (C is [M][N])
// perm==1 : C row = (m%T)*B + m/T    (C is [T][B][N], for x_proj time-major)
__global__ __launch_bounds__(256)
void gemm_bf16_wmma(const bf16_t* __restrict__ A, const bf16_t* __restrict__ Bt,
                    const float* __restrict__ bias, float* __restrict__ C,
                    int M, int N, int K, int perm) {
  int lane = threadIdx.x & 31;
  int wave = threadIdx.x >> 5;
  int unit = blockIdx.x * (blockDim.x >> 5) + wave;   // one 16(M)x64(N) strip
  int nblk = N >> 6;
  int mt   = unit / nblk, nb = unit % nblk;
  if (mt >= (M >> 4)) return;
  int m0 = mt << 4, n0 = nb << 6;

  v8f acc[4];
#pragma unroll
  for (int j = 0; j < 4; ++j) acc[j] = (v8f)0.0f;

  const bf16_t* arow = A + (size_t)m0 * K;
  const bf16_t* bp0  = Bt + (size_t)(n0 +  0) * K;
  const bf16_t* bp1  = Bt + (size_t)(n0 + 16) * K;
  const bf16_t* bp2  = Bt + (size_t)(n0 + 32) * K;
  const bf16_t* bp3  = Bt + (size_t)(n0 + 48) * K;

  // stage X prologue at kb=0
  v16bf aX  = load_A16(arow, K, 0, lane);
  v16bf bX0 = load_B16(bp0, K, 0, lane);
  v16bf bX1 = load_B16(bp1, K, 0, lane);
  v16bf bX2 = load_B16(bp2, K, 0, lane);
  v16bf bX3 = load_B16(bp3, K, 0, lane);

  for (int kb = 0; kb < K; kb += 64) {
    int k1 = kb + 32;                 // always < K (K multiple of 64)
    v16bf aY  = load_A16(arow, K, k1, lane);
    v16bf bY0 = load_B16(bp0, K, k1, lane);
    v16bf bY1 = load_B16(bp1, K, k1, lane);
    v16bf bY2 = load_B16(bp2, K, k1, lane);
    v16bf bY3 = load_B16(bp3, K, k1, lane);
    acc[0] = wmma_bf16(aX, bX0, acc[0]);
    acc[1] = wmma_bf16(aX, bX1, acc[1]);
    acc[2] = wmma_bf16(aX, bX2, acc[2]);
    acc[3] = wmma_bf16(aX, bX3, acc[3]);

    int k2 = (kb + 64) & (K - 1);     // wraps on last iteration (harmless)
    aX  = load_A16(arow, K, k2, lane);
    bX0 = load_B16(bp0, K, k2, lane);
    bX1 = load_B16(bp1, K, k2, lane);
    bX2 = load_B16(bp2, K, k2, lane);
    bX3 = load_B16(bp3, K, k2, lane);
    acc[0] = wmma_bf16(aY, bY0, acc[0]);
    acc[1] = wmma_bf16(aY, bY1, acc[1]);
    acc[2] = wmma_bf16(aY, bY2, acc[2]);
    acc[3] = wmma_bf16(aY, bY3, acc[3]);
  }

  int mrow = (lane >> 4) << 3;
#pragma unroll
  for (int j = 0; j < 4; ++j) {
    int ncol = n0 + 16 * j + (lane & 15);
    float bn2 = bias[ncol];
#pragma unroll
    for (int v = 0; v < 8; ++v) {
      int mg = m0 + mrow + v;
      size_t crow = perm ? ((size_t)(mg & (TT - 1)) * BB + (mg >> TSH)) : (size_t)mg;
      C[crow * N + ncol] = acc[j][v] + bn2;
    }
  }
}

// ---------------------------------------------------------------- persistent LSTM scan
// One workgroup (32 waves). Wave w owns hidden columns [16w,16w+16) for all 64 batches.
// h ping-pong in LDS; c lives in VGPRs; one barrier per timestep.
// xproj is time-major [T][B][4H]; step t+1's 512KB slab is prefetched during step t.
// Ut B-tiles: 4-register schedule -- bE/bO carry gates 0/1 loaded one kb ahead
// (wrapping across passes/timesteps), b2/b3 loaded per-kb for gates 2/3.
// All B loads land >=4 WMMAs before use; no rotation moves.
__global__ __launch_bounds__(1024)
void lstm_scan_kernel(const float* __restrict__ xproj,   // [T,B,4H] fp32
                      const bf16_t* __restrict__ Ut,     // [4H][H] bf16 (K-major)
                      float* __restrict__ out_hc,        // d_out (h_T,c_T at tail)
                      bf16_t* __restrict__ hid) {        // [B,T,H] bf16
  __shared__ __attribute__((aligned(16))) bf16_t hbuf[2][64 * HPAD];

  int lane = threadIdx.x & 31;
  int w    = threadIdx.x >> 5;            // 0..31
  int nloc = lane & 15;
  int mrow = (lane >> 4) << 3;            // 0 or 8
  int ncol = w << 4;                      // owned column base

  for (int i = threadIdx.x; i < 64 * HPAD; i += 1024) {
    hbuf[0][i] = (bf16_t)0.0f;
    hbuf[1][i] = (bf16_t)0.0f;
  }
  __syncthreads();

  v8f cst[4];
#pragma unroll
  for (int m = 0; m < 4; ++m) cst[m] = (v8f)0.0f;

  const bf16_t* Ug0 = Ut + (size_t)(0 * HH + ncol) * HH;
  const bf16_t* Ug1 = Ut + (size_t)(1 * HH + ncol) * HH;
  const bf16_t* Ug2 = Ut + (size_t)(2 * HH + ncol) * HH;
  const bf16_t* Ug3 = Ut + (size_t)(3 * HH + ncol) * HH;

  // rolling lookahead for gates 0/1 at kb=0 (sequence repeats every pass)
  v16bf bE = load_B16(Ug0, HH, 0, lane);
  v16bf bO = load_B16(Ug1, HH, 0, lane);

  int cur = 0;
  for (int t = 0; t < TT; ++t) {
    // Prefetch next timestep's x_proj slab (512KB contiguous) into L2/WGP$
    // while this step's WMMA chain runs: 4 x 128B lines per thread.
    if (t + 1 < TT) {
      const char* pf = (const char*)(xproj + (size_t)(t + 1) * BB * G4H)
                     + (size_t)threadIdx.x * 512;
      __builtin_prefetch(pf, 0, 3);
      __builtin_prefetch(pf + 128, 0, 3);
      __builtin_prefetch(pf + 256, 0, 3);
      __builtin_prefetch(pf + 384, 0, 3);
    }

    const bf16_t* hold = hbuf[cur];
    bf16_t*       hnew = hbuf[cur ^ 1];
    const float*  xpt  = xproj + (size_t)t * BB * G4H;   // [B][4H] slab for step t

    for (int p = 0; p < 2; ++p) {         // m-tile pairs {0,1} then {2,3}
      int mt0 = p << 1;
      v8f acc[4][2];
#pragma unroll
      for (int g = 0; g < 4; ++g) { acc[g][0] = (v8f)0.0f; acc[g][1] = (v8f)0.0f; }

      for (int kb = 0; kb < HH; kb += 32) {
        v16bf a0 = load_A16(hold + (size_t)(mt0 << 4) * HPAD,        HPAD, kb, lane);
        v16bf a1 = load_A16(hold + (size_t)((mt0 + 1) << 4) * HPAD,  HPAD, kb, lane);

        v16bf b2 = load_B16(Ug2, HH, kb, lane);          // gate 2 @ kb
        acc[0][0] = wmma_bf16(a0, bE, acc[0][0]);
        acc[0][1] = wmma_bf16(a1, bE, acc[0][1]);
        v16bf b3 = load_B16(Ug3, HH, kb, lane);          // gate 3 @ kb
        acc[1][0] = wmma_bf16(a0, bO, acc[1][0]);
        acc[1][1] = wmma_bf16(a1, bO, acc[1][1]);
        int kn = (kb + 32) & (HH - 1);                   // wraps to next pass/step
        bE = load_B16(Ug0, HH, kn, lane);                // gate 0 @ next kb
        acc[2][0] = wmma_bf16(a0, b2, acc[2][0]);
        acc[2][1] = wmma_bf16(a1, b2, acc[2][1]);
        bO = load_B16(Ug1, HH, kn, lane);                // gate 1 @ next kb
        acc[3][0] = wmma_bf16(a0, b3, acc[3][0]);
        acc[3][1] = wmma_bf16(a1, b3, acc[3][1]);
      }

#pragma unroll
      for (int mi = 0; mi < 2; ++mi) {
        int mt = mt0 + mi;
        int b0 = (mt << 4) + mrow;        // batch row base for this lane
#pragma unroll
        for (int v = 0; v < 8; ++v) {
          int brow = b0 + v;
          size_t rbase = (size_t)brow * G4H + ncol + nloc;
          float gi = acc[0][mi][v] + xpt[rbase + 0 * HH];
          float gf = acc[1][mi][v] + xpt[rbase + 1 * HH];
          float gg = acc[2][mi][v] + xpt[rbase + 2 * HH];
          float go = acc[3][mi][v] + xpt[rbase + 3 * HH];
          float it = sigmoidf_(gi);
          float ft = sigmoidf_(gf);
          float gt = tanhf(gg);
          float ot = sigmoidf_(go);
          float cn = ft * cst[mt][v] + it * gt;
          cst[mt][v] = cn;
          float hn = ot * tanhf(cn);
          hnew[(size_t)brow * HPAD + ncol + nloc] = (bf16_t)hn;
          hid[((size_t)brow * TT + t) * (size_t)HH + ncol + nloc] = (bf16_t)hn;
          if (t == TT - 1) {
            out_hc[OUT_H_OFF + (size_t)brow * HH + ncol + nloc] = hn;
            out_hc[OUT_C_OFF + (size_t)brow * HH + ncol + nloc] = cn;
          }
        }
      }
    }
    __syncthreads();                      // h_new visible to all before next step
    cur ^= 1;
  }
}

// ---------------------------------------------------------------- launch
extern "C" void kernel_launch(void* const* d_in, const int* in_sizes, int n_in,
                              void* d_out, int out_size, void* d_ws, size_t ws_size,
                              hipStream_t stream) {
  const float* x    = (const float*)d_in[0];   // [B,T,I]
  const float* W    = (const float*)d_in[1];   // [I,4H]
  const float* U    = (const float*)d_in[2];   // [H,4H]
  const float* bias = (const float*)d_in[3];   // [4H]
  const float* Wl   = (const float*)d_in[4];   // [H,O]
  const float* bl   = (const float*)d_in[5];   // [O]
  float* out = (float*)d_out;

  char* ws = (char*)d_ws;
  size_t off = 0;
  bf16_t* xb   = (bf16_t*)(ws + off); off += (size_t)BT * II * 2;      // 16 MB
  bf16_t* Wt   = (bf16_t*)(ws + off); off += (size_t)G4H * II * 2;     // 1 MB
  bf16_t* Ut   = (bf16_t*)(ws + off); off += (size_t)G4H * HH * 2;     // 2 MB
  bf16_t* Wlt  = (bf16_t*)(ws + off); off += (size_t)OO * HH * 2;      // 0.25 MB
  float*  xprj = (float*)(ws + off);  off += (size_t)BT * G4H * 4;     // 256 MB, [T,B,4H]
  bf16_t* hid  = (bf16_t*)(ws + off); off += (size_t)BT * HH * 2;      // 32 MB

  // pack / transpose to bf16 K-major operands
  pack_cast_bf16<<<8192, 256, 0, stream>>>(x, xb, BT * II);
  pack_transpose_bf16<<<(II * G4H + 255) / 256, 256, 0, stream>>>(W,  Wt,  II, G4H);
  pack_transpose_bf16<<<(HH * G4H + 255) / 256, 256, 0, stream>>>(U,  Ut,  HH, G4H);
  pack_transpose_bf16<<<(HH * OO  + 255) / 256, 256, 0, stream>>>(Wl, Wlt, HH, OO);

  // Phase 1: x_proj[T,B,4H] = x @ W + bias   (M=32768, N=2048, K=256), time-major store
  {
    int units = (BT >> 4) * (G4H >> 6);    // 65536 wave-strips
    gemm_bf16_wmma<<<units / 8, 256, 0, stream>>>(xb, Wt, bias, xprj, BT, G4H, II, 1);
  }

  // Phase 2: persistent sequential scan (writes hid, h_T, c_T)
  lstm_scan_kernel<<<1, 1024, 0, stream>>>(xprj, Ut, out, hid);

  // Phase 3: out = hidden_seq @ Wl + bl   (M=32768, N=256, K=512)
  {
    int units = (BT >> 4) * (OO >> 6);     // 8192 wave-strips
    gemm_bf16_wmma<<<units / 8, 256, 0, stream>>>(hid, Wlt, bl, out, BT, OO, HH, 0);
  }
}
